// GCN_LLM_12807592476696
// MI455X (gfx1250) — matrix operand order repaired and checked by reading, not compiled, not run on hardware
//
#include <hip/hip_runtime.h>
#include <hip/hip_bf16.h>
#include <stdint.h>

typedef __attribute__((ext_vector_type(16))) _Float16 v16h;
typedef __attribute__((ext_vector_type(8)))  float    v8f;
typedef int v4i_t __attribute__((vector_size(16)));                 // int __vector(4)
typedef __attribute__((address_space(1))) v4i_t* gv4i_p;            // global int4*
typedef __attribute__((address_space(3))) v4i_t* lv4i_p;            // LDS int4*

#define NN    262144
#define NE    1048576
#define NG    8192
#define F_IN  32
#define HID   128
#define LLMD  768
#define PROJD 10
#define EPSV  1e-5f

#if defined(__HIP_DEVICE_COMPILE__) && \
    __has_builtin(__builtin_amdgcn_global_load_async_to_lds_b128) && \
    __has_builtin(__builtin_amdgcn_s_wait_asynccnt)
#define HAS_ASYNC_LDS 1
#else
#define HAS_ASYNC_LDS 0
#endif

// ---------------- degree / normalization ----------------
__global__ void k_deg(const long long* __restrict__ dst, float* __restrict__ deg) {
    int e = blockIdx.x * blockDim.x + threadIdx.x;
    if (e < NE) atomicAdd(&deg[dst[e]], 1.0f);
}
__global__ void k_dis(float* __restrict__ deg) {
    int i = blockIdx.x * blockDim.x + threadIdx.x;
    if (i < NN) deg[i] = rsqrtf(1.0f + deg[i]);   // in-place: deg -> deg^{-1/2}
}

// ---------------- SMILES projection + gather ----------------
__global__ void k_proj(const float* __restrict__ smile, const float* __restrict__ Wp,
                       const float* __restrict__ bp, float* __restrict__ s) {
    int idx = blockIdx.x * blockDim.x + threadIdx.x;
    if (idx >= NG * PROJD) return;
    int g = idx / PROJD, p = idx % PROJD;
    float acc = bp[p];
    const float* row = smile + (size_t)g * LLMD;
    for (int k = 0; k < LLMD; ++k) acc = fmaf(row[k], Wp[k * PROJD + p], acc);
    s[idx] = fmaxf(acc, 0.0f);
}
__global__ void k_srep(const long long* __restrict__ batch, const float* __restrict__ s,
                       float* __restrict__ srep) {
    int idx = blockIdx.x * blockDim.x + threadIdx.x;
    if (idx >= NN * PROJD) return;
    int i = idx / PROJD, p = idx % PROJD;
    srep[idx] = s[(size_t)batch[i] * PROJD + p];
}

// ---------------- fragment packing (CDNA5 ISA layouts) ----------------
// A 16x32 f16: lanes 0-15 hold K={0..7,16..23}, lanes 16-31 hold K={8..15,24..31}
__device__ __forceinline__ size_t a_frag_idx(int r, int K, int nkt) {
    int kt = K >> 5, kk = K & 31;
    int rt = r >> 4, m = r & 15;
    int lane = m + ((kk & 8) ? 16 : 0);
    int fidx = (kk & 7) + ((kk & 16) ? 8 : 0);
    return (((size_t)rt * nkt + kt) * 32 + lane) * 16 + fidx;
}
__global__ void k_pack_a(const float* __restrict__ h, int hcols,
                         const float* __restrict__ srep, int scols,
                         int rows, int kpad, _Float16* __restrict__ A) {
    int idx = blockIdx.x * blockDim.x + threadIdx.x;
    int total = rows * kpad;
    if (idx >= total) return;
    int r = idx / kpad, K = idx % kpad;
    float val = 0.0f;
    if (K < hcols)              val = h[(size_t)r * hcols + K];
    else if (K < hcols + scols) val = srep[(size_t)r * PROJD + (K - hcols)];
    A[a_frag_idx(r, K, kpad >> 5)] = (_Float16)val;
}
// one-time: fill K=128..159 (srep + zero pad) of the 160-wide packed A
__global__ void k_pack_srep_tail(const float* __restrict__ srep, _Float16* __restrict__ A) {
    int idx = blockIdx.x * blockDim.x + threadIdx.x;
    if (idx >= NN * 32) return;
    int r = idx >> 5, j = idx & 31;
    float val = (j < PROJD) ? srep[(size_t)r * PROJD + j] : 0.0f;
    A[a_frag_idx(r, HID + j, 5)] = (_Float16)val;
}
// B 32x16 f16: lanes 0-15 hold K={0..7,16..23}, lanes 16-31 hold K={8..15,24..31}
__global__ void k_pack_w(const float* __restrict__ W, int kreal, int kpad,
                         _Float16* __restrict__ Wp) {
    int idx = blockIdx.x * blockDim.x + threadIdx.x;
    int total = kpad * HID;
    if (idx >= total) return;
    int K = idx / HID, n = idx % HID;
    int nkt = kpad >> 5;
    int kt = K >> 5, kk = K & 31;
    int nt = n >> 4, nn = n & 15;
    int lane = nn + ((kk & 8) ? 16 : 0);
    int fidx = (kk & 7) + ((kk & 16) ? 8 : 0);
    float v = (K < kreal) ? W[(size_t)K * HID + n] : 0.0f;
    Wp[(((size_t)nt * nkt + kt) * 32 + lane) * 16 + fidx] = (_Float16)v;
}

// ---------------- WMMA GEMM ----------------
// block = 8 waves = 64 rows x 128 cols. Wave w owns N-tile w and 4 M-tiles.
// A fragments staged through LDS (async-to-LDS when available).
// Epilogue optionally fuses GCN self-loop term + bias: hout = lin*dis^2 + bias.
__global__ void __launch_bounds__(256)
k_gemm_wmma(const _Float16* __restrict__ A, const _Float16* __restrict__ W,
            float* __restrict__ lin, float* __restrict__ hout,
            const float* __restrict__ dis, const float* __restrict__ bias, int nkt) {
    __shared__ _Float16 As[4 * 32 * 16];           // 4 KB: 4 M-tiles of A fragments
    const int rt0  = blockIdx.x * 4;
    const int tid  = threadIdx.x;
    const int wave = tid >> 5;                      // N-tile 0..7
    const int lane = tid & 31;
    // staging role: 256 threads x 16B = 4KB per K-step
    const int smt   = tid >> 6;                     // M-tile 0..3
    const int slane = (tid & 63) >> 1;              // lane 0..31
    const int shalf = tid & 1;                      // which 8-half chunk
    const int lbase = (smt * 32 + slane) * 16 + shalf * 8;

    const v16h* __restrict__ Wv  = (const v16h*)W;
    const v16h* __restrict__ Ls  = (const v16h*)As;
    v8f acc[4] = {{}, {}, {}, {}};

    for (int kt = 0; kt < nkt; ++kt) {
        __syncthreads();                            // previous tile consumed
        size_t gbase = (((size_t)(rt0 + smt) * nkt + kt) * 32 + slane) * 16 + (size_t)shalf * 8;
#if HAS_ASYNC_LDS
        __builtin_amdgcn_global_load_async_to_lds_b128(
            (gv4i_p)(void*)(A + gbase),
            (lv4i_p)(void*)&As[lbase], 0, 0);
        __builtin_amdgcn_s_wait_asynccnt(0);
#else
        *(uint4*)&As[lbase] = *(const uint4*)(A + gbase);
#endif
        __syncthreads();                            // tile staged
        v16h b = Wv[((size_t)wave * nkt + kt) * 32 + lane];
#pragma unroll
        for (int mt = 0; mt < 4; ++mt) {
            v16h a = Ls[mt * 32 + lane];
            acc[mt] = __builtin_amdgcn_wmma_f32_16x16x32_f16(
                false, a, false, b, (short)0, acc[mt], false, false);
        }
    }
    // C/D layout: VGPR v -> M=v (lanes 0-15) / M=v+8 (lanes 16-31), N=lane&15
    const int m0  = (lane < 16) ? 0 : 8;
    const int col = wave * 16 + (lane & 15);
    if (hout) {
        float bcol = bias[col];
#pragma unroll
        for (int mt = 0; mt < 4; ++mt) {
            size_t row0 = (size_t)(rt0 + mt) * 16 + m0;
#pragma unroll
            for (int v = 0; v < 8; ++v) {
                size_t r = row0 + v;
                float val = acc[mt][v];
                lin[r * HID + col] = val;
                float d = dis[r];
                hout[r * HID + col] = val * d * d + bcol;
            }
        }
    } else {
#pragma unroll
        for (int mt = 0; mt < 4; ++mt) {
            size_t row0 = (size_t)(rt0 + mt) * 16 + m0;
#pragma unroll
            for (int v = 0; v < 8; ++v)
                lin[(row0 + v) * HID + col] = acc[mt][v];
        }
    }
}

// ---------------- GCN edge aggregation ----------------
__global__ void k_edge_agg(const long long* __restrict__ src, const long long* __restrict__ dst,
                           const float* __restrict__ dis, const float* __restrict__ lin,
                           float* __restrict__ out) {
    int e = blockIdx.x;
    long long sN = src[e], dN = dst[e];
    float norm = dis[sN] * dis[dN];
    int c = threadIdx.x;
    atomicAdd(&out[(size_t)dN * HID + c], lin[(size_t)sN * HID + c] * norm);
}

// ---------------- BatchNorm over relu(h) ----------------
__global__ void k_bn_stats(const float* __restrict__ h, float* __restrict__ stats) {
    int c = threadIdx.x;                       // 0..127
    int rowsPer = NN / gridDim.x;
    size_t r0 = (size_t)blockIdx.x * rowsPer;
    float s = 0.0f, q = 0.0f;
    for (int r = 0; r < rowsPer; ++r) {
        float v = fmaxf(h[(r0 + r) * HID + c], 0.0f);
        s += v; q += v * v;
    }
    atomicAdd(&stats[c], s);
    atomicAdd(&stats[HID + c], q);
}
// normalizes in place; optionally also writes packed f16 A fragments (160-wide layout)
__global__ void k_bn_apply(float* __restrict__ h, const float* __restrict__ stats,
                           const float* __restrict__ gamma, const float* __restrict__ beta,
                           _Float16* __restrict__ Apack) {
    int idx = blockIdx.x * blockDim.x + threadIdx.x;
    if (idx >= NN * HID) return;
    int r = idx >> 7, c = idx & 127;
    float m   = stats[c] * (1.0f / NN);
    float var = stats[HID + c] * (1.0f / NN) - m * m;
    float v = fmaxf(h[idx], 0.0f);
    float o = (v - m) * rsqrtf(var + EPSV) * gamma[c] + beta[c];
    h[idx] = o;
    if (Apack) Apack[a_frag_idx(r, c, 5)] = (_Float16)o;
}

// ---------------- mean pooling ----------------
__global__ void k_counts(const long long* __restrict__ batch, float* __restrict__ counts) {
    int i = blockIdx.x * blockDim.x + threadIdx.x;
    if (i < NN) atomicAdd(&counts[batch[i]], 1.0f);
}
__global__ void k_pool(const long long* __restrict__ batch, const float* __restrict__ h,
                       float* __restrict__ pooled) {
    int i = blockIdx.x;
    int c = threadIdx.x;
    atomicAdd(&pooled[(size_t)batch[i] * HID + c], h[(size_t)i * HID + c]);
}
__global__ void k_pool_div(float* __restrict__ pooled, const float* __restrict__ counts) {
    int idx = blockIdx.x * blockDim.x + threadIdx.x;
    if (idx >= NG * HID) return;
    pooled[idx] /= fmaxf(counts[idx >> 7], 1.0f);
}

// ---------------- head: out = relu(z + bl1) @ Wl2 + bl2 ----------------
__global__ void k_head(const float* __restrict__ z, const float* __restrict__ bl1,
                       const float* __restrict__ Wl2, const float* __restrict__ bl2,
                       float* __restrict__ out) {
    int idx = blockIdx.x * blockDim.x + threadIdx.x;
    if (idx >= NG * 2) return;
    int g = idx >> 1, j = idx & 1;
    float acc = bl2[j];
    for (int c = 0; c < HID; ++c)
        acc = fmaf(fmaxf(z[(size_t)g * HID + c] + bl1[c], 0.0f), Wl2[c * 2 + j], acc);
    out[idx] = acc;
}

extern "C" void kernel_launch(void* const* d_in, const int* in_sizes, int n_in,
                              void* d_out, int out_size, void* d_ws, size_t ws_size,
                              hipStream_t stream) {
    (void)in_sizes; (void)n_in; (void)out_size; (void)ws_size;
    const float*     x     = (const float*)d_in[0];
    const long long* ei    = (const long long*)d_in[1];   // [2, NE] int64
    const long long* batch = (const long long*)d_in[2];   // [NN] int64
    const float*     smile = (const float*)d_in[3];
    const float*     Wp    = (const float*)d_in[4];
    const float*     bp    = (const float*)d_in[5];
    const float*     W1    = (const float*)d_in[6];
    const float*     b1    = (const float*)d_in[7];
    const float*     Wc    = (const float*)d_in[8];       // [5,138,128]
    const float*     bc    = (const float*)d_in[9];       // [5,128]
    const float*     gamma = (const float*)d_in[10];      // [6,128]
    const float*     beta  = (const float*)d_in[11];
    const float*     Wl1   = (const float*)d_in[12];
    const float*     bl1   = (const float*)d_in[13];
    const float*     Wl2   = (const float*)d_in[14];
    const float*     bl2   = (const float*)d_in[15];
    const long long* src = ei;
    const long long* dst = ei + NE;

    // carve workspace
    char* base = (char*)d_ws;
    size_t off = 0;
    auto carve = [&](size_t bytes) -> char* {
        char* p = base + off;
        off += (bytes + 255) & ~(size_t)255;
        return p;
    };
    float*     dis    = (float*)    carve((size_t)NN * 4);
    float*     sproj  = (float*)    carve((size_t)NG * PROJD * 4);
    float*     srep   = (float*)    carve((size_t)NN * PROJD * 4);
    _Float16*  A16a   = (_Float16*) carve((size_t)NN * 64 * 2);    // layer-1 (also head reuse)
    _Float16*  A16b   = (_Float16*) carve((size_t)NN * 160 * 2);   // layers 2-6
    _Float16*  W16    = (_Float16*) carve((size_t)160 * HID * 2);
    float*     lin    = (float*)    carve((size_t)NN * HID * 4);
    float*     h      = (float*)    carve((size_t)NN * HID * 4);
    float*     stats  = (float*)    carve((size_t)2 * HID * 4);
    float*     pooled = (float*)    carve((size_t)NG * HID * 4);
    float*     counts = (float*)    carve((size_t)NG * 4);
    float*     z      = (float*)    carve((size_t)NG * HID * 4);

    const int T = 256;
    // degree -> dis
    (void)hipMemsetAsync(dis, 0, (size_t)NN * 4, stream);
    k_deg<<<(NE + T - 1) / T, T, 0, stream>>>(dst, dis);
    k_dis<<<(NN + T - 1) / T, T, 0, stream>>>(dis);
    // SMILES projection + per-node gather; pre-fill invariant tail of packed A
    k_proj<<<(NG * PROJD + T - 1) / T, T, 0, stream>>>(smile, Wp, bp, sproj);
    k_srep<<<(NN * PROJD + T - 1) / T, T, 0, stream>>>(batch, sproj, srep);
    k_pack_srep_tail<<<(NN * 32 + T - 1) / T, T, 0, stream>>>(srep, A16b);

    // ---- layer 1: [x|srep] (K=42 -> pad 64) ----
    k_pack_a<<<(NN * 64 + T - 1) / T, T, 0, stream>>>(x, F_IN, srep, PROJD, NN, 64, A16a);
    k_pack_w<<<(64 * HID + T - 1) / T, T, 0, stream>>>(W1, F_IN + PROJD, 64, W16);
    k_gemm_wmma<<<NN / 64, 256, 0, stream>>>(A16a, W16, lin, h, dis, b1, 2);
    k_edge_agg<<<NE, HID, 0, stream>>>(src, dst, dis, lin, h);
    (void)hipMemsetAsync(stats, 0, 2 * HID * 4, stream);
    k_bn_stats<<<1024, HID, 0, stream>>>(h, stats);
    k_bn_apply<<<(NN * HID + T - 1) / T, T, 0, stream>>>(h, stats, gamma, beta, A16b);

    // ---- layers 2..6: [h|srep] (K=138 -> pad 160); h-part packed by bn_apply ----
    for (int i = 0; i < 5; ++i) {
        k_pack_w<<<(160 * HID + T - 1) / T, T, 0, stream>>>(Wc + (size_t)i * 138 * HID, 138, 160, W16);
        k_gemm_wmma<<<NN / 64, 256, 0, stream>>>(A16b, W16, lin, h, dis, bc + (size_t)i * HID, 5);
        k_edge_agg<<<NE, HID, 0, stream>>>(src, dst, dis, lin, h);
        (void)hipMemsetAsync(stats, 0, 2 * HID * 4, stream);
        k_bn_stats<<<1024, HID, 0, stream>>>(h, stats);
        k_bn_apply<<<(NN * HID + T - 1) / T, T, 0, stream>>>(
            h, stats, gamma + (size_t)(i + 1) * HID, beta + (size_t)(i + 1) * HID,
            (i < 4) ? A16b : (_Float16*)nullptr);
    }

    // ---- mean pool per graph ----
    (void)hipMemsetAsync(pooled, 0, (size_t)NG * HID * 4, stream);
    (void)hipMemsetAsync(counts, 0, (size_t)NG * 4, stream);
    k_counts<<<(NN + T - 1) / T, T, 0, stream>>>(batch, counts);
    k_pool<<<NN, HID, 0, stream>>>(batch, h, pooled);
    k_pool_div<<<(NG * HID + T - 1) / T, T, 0, stream>>>(pooled, counts);

    // ---- head: pooled @ Wl1 via WMMA, then tiny 128->2 ----
    k_pack_a<<<(NG * HID + T - 1) / T, T, 0, stream>>>(pooled, HID, (const float*)nullptr, 0, NG, 128, A16a);
    k_pack_w<<<(128 * HID + T - 1) / T, T, 0, stream>>>(Wl1, HID, 128, W16);
    k_gemm_wmma<<<NG / 64, 256, 0, stream>>>(A16a, W16, z, (float*)nullptr,
                                             (const float*)nullptr, (const float*)nullptr, 4);
    k_head<<<(NG * 2 + T - 1) / T, T, 0, stream>>>(z, bl1, Wl2, bl2, (float*)d_out);
}
